// TokenChoiceTopKRouter_32701880992123
// MI455X (gfx1250) — matrix operand order repaired and checked by reading, not compiled, not run on hardware
//
#include <hip/hip_runtime.h>
#include <hip/hip_bf16.h>

#define DIM        4096
#define NEXP       64
#define TOPK       8
#define KC         64     // K-chunk staged in LDS per iteration
#define TOK_PER_WG 128    // 8 waves x 16 tokens
#define THREADS    256

typedef __attribute__((ext_vector_type(16))) __bf16 v16bf;
typedef __attribute__((ext_vector_type(8)))  __bf16 v8bf;
typedef __attribute__((ext_vector_type(4)))  __bf16 v4bf;
typedef __attribute__((ext_vector_type(4)))  float  v4f;
typedef __attribute__((ext_vector_type(8)))  float  v8f;

// convert float4 -> 4x bf16 (vector fptrunc, RNE) and store 8 bytes to LDS
static __device__ __forceinline__ void store_bf16x4(unsigned short* d, const float4 v) {
  v4f f = { v.x, v.y, v.z, v.w };
  v4bf t = __builtin_convertvector(f, v4bf);
  *(v4bf*)d = t;
}

__global__ void init_counts_kernel(float* __restrict__ counts) {
  if (threadIdx.x < NEXP) counts[threadIdx.x] = 0.0f;
}

__global__ void __launch_bounds__(THREADS)
router_kernel(const float* __restrict__ x, const float* __restrict__ gw,
              const float* __restrict__ eb, float* __restrict__ out_scores,
              float* __restrict__ out_idx, float* __restrict__ counts) {
  __shared__ unsigned short xs[2][TOK_PER_WG][KC];  // 32 KB bf16 x tiles (double buf)
  __shared__ unsigned short ws[2][NEXP][KC];        // 16 KB bf16 gate_w tiles
  __shared__ float sc[8][16][NEXP + 1];             // ~33 KB sigmoid scores, padded
  __shared__ float bsh[NEXP];

  const int tid  = threadIdx.x;
  const int lane = tid & 31;
  const int wave = tid >> 5;
  const int m    = lane & 15;   // column / row-in-tile index
  const int half = lane >> 4;   // lane half selects K sub-range (ISA bf16 layout)
  const size_t tok0 = (size_t)blockIdx.x * TOK_PER_WG;

  if (tid < NEXP) bsh[tid] = eb[tid];

  // staging geometry: x -> 8 float4/thread, gate_w -> 4 float4/thread
  const int xr = tid >> 4;         // 0..15
  const int xc = (tid & 15) * 4;   // 0..60

  // ---- preload K-chunk 0 into registers ----
  float4 rx[8], rw[4];
  #pragma unroll
  for (int p = 0; p < 8; ++p)
    rx[p] = *(const float4*)(x + (tok0 + p * 16 + xr) * DIM + xc);
  #pragma unroll
  for (int p = 0; p < 4; ++p) {
    const int q = p * THREADS + tid;
    rw[p] = *(const float4*)(gw + (size_t)(q >> 4) * DIM + (q & 15) * 4);
  }

  v8f acc[4] = {};  // 16 tokens x 64 experts, f32 accumulate
  int buf = 0;

  for (int kc = 0; kc < DIM; kc += KC) {
    // ---- convert + store staged regs into LDS[buf] ----
    #pragma unroll
    for (int p = 0; p < 8; ++p)
      store_bf16x4(&xs[buf][p * 16 + xr][xc], rx[p]);
    #pragma unroll
    for (int p = 0; p < 4; ++p) {
      const int q = p * THREADS + tid;
      store_bf16x4(&ws[buf][q >> 4][(q & 15) * 4], rw[p]);
    }
    // ---- issue next chunk's global loads (overlap with barrier + WMMA) ----
    if (kc + KC < DIM) {
      #pragma unroll
      for (int p = 0; p < 8; ++p)
        rx[p] = *(const float4*)(x + (tok0 + p * 16 + xr) * DIM + kc + KC + xc);
      #pragma unroll
      for (int p = 0; p < 4; ++p) {
        const int q = p * THREADS + tid;
        rw[p] = *(const float4*)(gw + (size_t)(q >> 4) * DIM + kc + KC + (q & 15) * 4);
      }
    }
    __syncthreads();  // single barrier per chunk (double-buffered LDS)

    // ---- WMMA: 16 tokens x 64 experts, K=64 in two 32-steps ----
    const int tr = wave * 16 + m;  // A-fragment row (token within WG tile)
    #pragma unroll
    for (int ks = 0; ks < KC; ks += 32) {
      // A 16x32 bf16: lane half 0 holds K[0..8)+[16..24), half 1 K[8..16)+[24..32)
      union { v16bf v; v8bf h[2]; } a;
      a.h[0] = *(const v8bf*)&xs[buf][tr][ks + half * 8];
      a.h[1] = *(const v8bf*)&xs[buf][tr][ks + 16 + half * 8];
      #pragma unroll
      for (int g = 0; g < 4; ++g) {
        // B 32x16 bf16: lane = column (expert), half 0 holds K[0..8)+[16..24), half 1 K[8..16)+[24..32)
        union { v16bf v; v8bf h[2]; } b;
        const unsigned short* bp = &ws[buf][g * 16 + m][ks + half * 8];
        b.h[0] = *(const v8bf*)(bp);
        b.h[1] = *(const v8bf*)(bp + 16);
        acc[g] = __builtin_amdgcn_wmma_f32_16x16x32_bf16(
            false, a.v, false, b.v, (short)0, acc[g], false, false);
      }
    }
    buf ^= 1;
  }

  // ---- sigmoid + scatter: C/D layout => VGPR j, lane L: M=j+8*(L/16), N=L%16 ----
  #pragma unroll
  for (int g = 0; g < 4; ++g)
    #pragma unroll
    for (int j = 0; j < 8; ++j) {
      const int t = j + half * 8;
      const int e = g * 16 + m;
      sc[wave][t][e] = 1.0f / (1.0f + __expf(-acc[g][j]));
    }
  __syncthreads();

  // ---- top-8: lane pair (t, t+16) splits the 64-expert scan ----
  {
    const int t    = lane & 15;
    const int side = lane >> 4;           // 0: experts [0,32), 1: [32,64)
    const size_t gtok = tok0 + wave * 16 + t;
    float vals[TOPK]; int inds[TOPK];
    float ssum = 0.0f;
    #pragma unroll
    for (int k = 0; k < TOPK; ++k) {
      float best = -__builtin_inff(); int bi = side * 32;
      for (int e2 = 0; e2 < 32; ++e2) {
        const int e = side * 32 + e2;
        const float v = sc[wave][t][e] + bsh[e];
        if (v > best) { best = v; bi = e; }      // '>' keeps lowest index on ties
      }
      const float ob = __shfl_xor(best, 16, 32); // combine halves
      const int   oi = __shfl_xor(bi, 16, 32);
      if (ob > best || (ob == best && oi < bi)) { best = ob; bi = oi; }
      const float raw = sc[wave][t][bi];
      vals[k] = raw; inds[k] = bi; ssum += raw;
      sc[wave][t][bi] = -__builtin_inff();       // both pair lanes write same value
    }
    const float inv = 1.0f / (ssum + 1e-20f);    // ROUTE_SCALE == 1.0
    if (side == 0) {
      #pragma unroll
      for (int k = 0; k < TOPK; ++k) {
        out_scores[gtok * TOPK + k] = vals[k] * inv;
        out_idx[gtok * TOPK + k]    = (float)inds[k];
        atomicAdd(&counts[inds[k]], 1.0f);       // exact: integer-valued f32 adds
      }
    }
  }
}

extern "C" void kernel_launch(void* const* d_in, const int* in_sizes, int n_in,
                              void* d_out, int out_size, void* d_ws, size_t ws_size,
                              hipStream_t stream) {
  (void)n_in; (void)out_size; (void)d_ws; (void)ws_size;
  const float* x  = (const float*)d_in[0];
  const float* gw = (const float*)d_in[1];
  const float* eb = (const float*)d_in[2];
  const int n_tokens = in_sizes[0] / DIM;  // 16384

  float* out        = (float*)d_out;
  float* out_scores = out;                                   // [N,8]
  float* out_idx    = out + (size_t)n_tokens * TOPK;         // [N,8] (indices as f32)
  float* counts     = out + (size_t)n_tokens * TOPK * 2;     // [64]

  init_counts_kernel<<<1, 64, 0, stream>>>(counts);
  router_kernel<<<n_tokens / TOK_PER_WG, THREADS, 0, stream>>>(
      x, gw, eb, out_scores, out_idx, counts);
}
